// Spatial_Attention_58110907515327
// MI455X (gfx1250) — matrix operand order repaired and checked by reading, not compiled, not run on hardware
//
#include <hip/hip_runtime.h>

typedef _Float16 h16;
typedef __attribute__((ext_vector_type(16))) _Float16 v16h;
typedef __attribute__((ext_vector_type(8)))  _Float16 v8h;
typedef __attribute__((ext_vector_type(8)))  float    v8f;

#define TOK   17
#define TOKP  32
#define CH    256
#define NH    8
#define HDIM  32
#define QK_SCALE 0.17677669529663687f   // 1/sqrt(32)

// ---------------- CDNA5 helpers ----------------

__device__ __forceinline__ void lds_fence() {
  // per-wave LDS store->load visibility (cross-lane); DS ops are in-order,
  // this waits for outstanding DS returns.
  asm volatile("s_wait_dscnt 0" ::: "memory");
}

__device__ __forceinline__ v8f wmma16(v16h a, v16h b, v8f c) {
  return __builtin_amdgcn_wmma_f32_16x16x32_f16(false, a, false, b, (short)0, c,
                                                false, false);
}

// A fragment (16x32 f16) from a row-major f16 buffer (LDS). Lane layout per
// ISA: lane<16 row=lane, K {0..7,16..23}; lane>=16 same rows, K {8..15,24..31}.
__device__ __forceinline__ v16h loadA(const h16* base, int pitch, int rowBase,
                                      int kBase, int lane) {
  const int r  = lane & 15;
  const int ks = (lane >> 4) << 3;            // 0 or 8
  const h16* p = base + (size_t)(rowBase + r) * pitch + kBase + ks;
  v8h lo = *(const v8h*)p;
  v8h hi = *(const v8h*)(p + 16);
  return __builtin_shufflevector(lo, hi, 0, 1, 2, 3, 4, 5, 6, 7, 8, 9, 10, 11,
                                 12, 13, 14, 15);
}

// B fragment (32x16 f16) from a TRANSPOSED ([n][k] row-major) f16 buffer.
// lane<16 col=lane, K {0..7,16..23}; lane>=16 col=lane-16, K {8..15,24..31}.
__device__ __forceinline__ v16h loadB(const h16* baseT, int pitch, int nBase,
                                      int kBase, int lane) {
  const int n  = lane & 15;
  const int kk = (lane >> 4) << 3;            // 0 or 8
  const h16* p = baseT + (size_t)(nBase + n) * pitch + kBase + kk;
  v8h lo = *(const v8h*)p;
  v8h hi = *(const v8h*)(p + 16);
  return __builtin_shufflevector(lo, hi, 0, 1, 2, 3, 4, 5, 6, 7, 8, 9, 10, 11,
                                 12, 13, 14, 15);
}

// Pack 16 fp32 (two 32B runs) into an A-layout f16 fragment.
__device__ __forceinline__ v16h cvtFrag(float4 a, float4 b, float4 c,
                                        float4 d) {
  v16h v;
  v[0] = (h16)a.x;  v[1] = (h16)a.y;  v[2] = (h16)a.z;  v[3] = (h16)a.w;
  v[4] = (h16)b.x;  v[5] = (h16)b.y;  v[6] = (h16)b.z;  v[7] = (h16)b.w;
  v[8] = (h16)c.x;  v[9] = (h16)c.y;  v[10] = (h16)c.z; v[11] = (h16)c.w;
  v[12] = (h16)d.x; v[13] = (h16)d.y; v[14] = (h16)d.z; v[15] = (h16)d.w;
  return v;
}

// Store a 16x16 f32 D fragment as f16 into a row-major LDS tile.
// D layout: VGPR r -> (M = r + 8*(lane>=16), N = lane&15).
__device__ __forceinline__ void storeD(h16* base, int pitch, int rowBase,
                                       int colBase, int lane, v8f acc) {
  const int c  = colBase + (lane & 15);
  const int r0 = rowBase + ((lane >> 4) << 3);
#pragma unroll
  for (int r = 0; r < 8; ++r) base[(r0 + r) * pitch + c] = (h16)acc[r];
}

// Same but transposed (for building V^T).
__device__ __forceinline__ void storeDT(h16* base, int pitch, int rowBase,
                                        int colBase, int lane, v8f acc) {
  const int c  = colBase + (lane & 15);
  const int r0 = rowBase + ((lane >> 4) << 3);
#pragma unroll
  for (int r = 0; r < 8; ++r) base[c * pitch + (r0 + r)] = (h16)acc[r];
}

// ------- weight prep: fp32 [K][N] -> f16 [N][K]; W_gate -> f16 copy -------

__global__ void prep_weights(const float* __restrict__ Wqkv,
                             const float* __restrict__ Wproj,
                             const float* __restrict__ Wgate,
                             h16* __restrict__ wqkvT,
                             h16* __restrict__ wprojT,
                             h16* __restrict__ wgF) {
  int idx = blockIdx.x * blockDim.x + threadIdx.x;
  if (idx < 768 * 256) {
    int n = idx >> 8, k = idx & 255;
    wqkvT[idx] = (h16)Wqkv[k * 768 + n];
  } else if (idx < 768 * 256 + 256 * 256) {
    int i = idx - 768 * 256;
    int n = i >> 8, k = i & 255;
    wprojT[i] = (h16)Wproj[k * 256 + n];
  } else if (idx < 768 * 256 + 256 * 256 + 256) {
    int i = idx - (768 * 256 + 256 * 256);
    wgF[i] = (h16)Wgate[i];
  }
}

// ---------------- fused attention kernel ----------------
// 4 waves/block, one batch element per wave. x lives in registers as WMMA
// A-fragments (ks loops FULLY unrolled so the array is SROA-promoted to
// VGPRs); qkv inner GEMM is global B-fragment loads + wmma only.

#define PW_BYTES 22528                 // ao 16K + q/k/vt 3*2K
#define SM_BYTES (3072 + 4 * PW_BYTES) // 93184

__global__ __launch_bounds__(128, 1)
void fused_attn(const float* __restrict__ x, const float* __restrict__ adj,
                const float* __restrict__ bgate, const float* __restrict__ bproj,
                const h16* __restrict__ wqkvT, const h16* __restrict__ wprojT,
                const h16* __restrict__ wgF, float* __restrict__ out) {
  extern __shared__ char smem[];
  const int tid  = threadIdx.x;
  const int lane = tid & 31;
  const int wid  = tid >> 5;
  const int b    = blockIdx.x * 4 + wid;

  float* adjS    = (float*)smem;                 // 289 f32 (pad 1280B)
  float* bpS     = (float*)(smem + 1280);        // 256 f32
  float* gateAll = (float*)(smem + 2304);        // 4*32 f32
  char*  wb      = smem + 3072 + wid * PW_BYTES;
  h16* ao  = (h16*)wb;                           // [32][256] f16 attn out
  h16* sq  = (h16*)(wb + 16384);                 // [32][32] Q, then P
  h16* sk  = (h16*)(wb + 18432);                 // [32][32] K
  h16* svt = (h16*)(wb + 20480);                 // [32][32] V^T ([d][key])
  float* gateW = gateAll + wid * 32;

  // ---- block prologue: tiny shared tables ----
  for (int i = tid; i < TOK * TOK; i += 128) adjS[i] = adj[i];
  for (int i = tid; i < CH; i += 128) bpS[i] = bproj[i];
  __syncthreads();

  // ---- load x straight into register A-fragments (fp32 -> f16) ----
  // aX[mt][ks] covers rows mt*16..mt*16+15, K = ks*32..ks*32+31.
  const int r   = lane & 15;
  const int ks8 = (lane >> 4) << 3;              // 0 or 8
  v16h aX[2][8];
#pragma unroll
  for (int mt = 0; mt < 2; ++mt) {
    const int row = mt * 16 + r;
    if (row < TOK) {
      const float* rp = x + ((size_t)b * TOK + row) * CH;
#pragma unroll
      for (int ks = 0; ks < 8; ++ks) {
        const float* p = rp + ks * 32 + ks8;
        aX[mt][ks] = cvtFrag(*(const float4*)p, *(const float4*)(p + 4),
                             *(const float4*)(p + 16), *(const float4*)(p + 20));
      }
    } else {
      v16h z = {};
#pragma unroll
      for (int ks = 0; ks < 8; ++ks) aX[mt][ks] = z;
    }
  }

  // ---- key gate: sigmoid(x @ W_gate + b_gate) from register fragments ----
  {
    const float bg = bgate[0];
    float part0 = 0.f, part1 = 0.f;
#pragma unroll
    for (int ks = 0; ks < 8; ++ks) {
      const h16* p = wgF + ks * 32 + ks8;        // same k-mapping as aX
      v8h wlo = *(const v8h*)p;
      v8h whi = *(const v8h*)(p + 16);
      v16h w = __builtin_shufflevector(wlo, whi, 0, 1, 2, 3, 4, 5, 6, 7, 8, 9,
                                       10, 11, 12, 13, 14, 15);
#pragma unroll
      for (int j = 0; j < 16; ++j) {
        part0 += (float)aX[0][ks][j] * (float)w[j];
        part1 += (float)aX[1][ks][j] * (float)w[j];
      }
    }
    // lanes L and L+16 hold complementary k-halves of the same row
    part0 += __shfl_xor(part0, 16, 32);
    part1 += __shfl_xor(part1, 16, 32);
    if (lane < 16) {
      gateW[lane]      = 1.f / (1.f + __expf(-(part0 + bg)));
      gateW[16 + lane] = 1.f / (1.f + __expf(-(part1 + bg)));
    }
  }
  lds_fence();

  const v8f zero = {};

  // ---- per-head: qkv slices -> gated scores -> softmax -> P@V ----
#pragma unroll 1
  for (int h = 0; h < NH; ++h) {
    // Q, K, V head tiles [32 tok][32 d]; V stored transposed.
#pragma unroll 1
    for (int s = 0; s < 3; ++s) {
      h16* dst = (s == 0) ? sq : (s == 1) ? sk : svt;
#pragma unroll 1
      for (int nt = 0; nt < 2; ++nt) {
        const int col = s * CH + h * HDIM + nt * 16;   // qkv column base
        v8f acc0 = zero, acc1 = zero;
        // FULL unroll: aX indices must be compile-time constants so the
        // fragments stay in VGPRs (no scratch).
#pragma unroll
        for (int ks = 0; ks < 8; ++ks) {
          v16h bf = loadB(wqkvT, CH, col, ks * 32, lane);
          acc0 = wmma16(aX[0][ks], bf, acc0);
          acc1 = wmma16(aX[1][ks], bf, acc1);
        }
        if (s < 2) {
          storeD(dst, TOKP, 0, nt * 16, lane, acc0);
          storeD(dst, TOKP, 16, nt * 16, lane, acc1);
        } else {
          storeDT(dst, TOKP, 0, nt * 16, lane, acc0);
          storeDT(dst, TOKP, 16, nt * 16, lane, acc1);
        }
      }
    }
    lds_fence();

    // scores = Q K^T   (A from sq [q][d], B from sk [key][d] == transposed)
    v8f sc[2][2];
    {
      v16h a0 = loadA(sq, TOKP, 0, 0, lane);
      v16h a1 = loadA(sq, TOKP, 16, 0, lane);
      v16h b0 = loadB(sk, TOKP, 0, 0, lane);
      v16h b1 = loadB(sk, TOKP, 16, 0, lane);
      sc[0][0] = wmma16(a0, b0, zero);
      sc[0][1] = wmma16(a0, b1, zero);
      sc[1][0] = wmma16(a1, b0, zero);
      sc[1][1] = wmma16(a1, b1, zero);
    }

    // gate-blend + masked softmax over keys, write P (f16) into sq.
    {
      const int key0 = lane & 15;
      const int key1 = 16 + key0;
      const float g0 = gateW[key0];
      const float g1 = (key1 < TOK) ? gateW[key1] : 0.f;
#pragma unroll
      for (int mt = 0; mt < 2; ++mt) {
#pragma unroll
        for (int rr = 0; rr < 8; ++rr) {
          const int qt = mt * 16 + ((lane >> 4) << 3) + rr;
          const int qa = qt > 16 ? 16 : qt;      // adj row clamp (pads unused)
          float v0 = sc[mt][0][rr] * QK_SCALE;
          float v1 = sc[mt][1][rr] * QK_SCALE;
          v0 = v0 * (1.f - g0) + adjS[qa * TOK + key0] * g0;
          v1 = (key1 < TOK) ? (v1 * (1.f - g1) + adjS[qa * TOK + key1] * g1)
                            : -3.0e38f;
          float m = fmaxf(v0, v1);
          m = fmaxf(m, __shfl_xor(m, 1, 32));
          m = fmaxf(m, __shfl_xor(m, 2, 32));
          m = fmaxf(m, __shfl_xor(m, 4, 32));
          m = fmaxf(m, __shfl_xor(m, 8, 32));
          float e0 = __expf(v0 - m);
          float e1 = (key1 < TOK) ? __expf(v1 - m) : 0.f;
          float sum = e0 + e1;
          sum += __shfl_xor(sum, 1, 32);
          sum += __shfl_xor(sum, 2, 32);
          sum += __shfl_xor(sum, 4, 32);
          sum += __shfl_xor(sum, 8, 32);
          const float inv = 1.f / sum;
          sq[qt * TOKP + key0] = (h16)(e0 * inv);
          sq[qt * TOKP + key1] = (h16)(e1 * inv);
        }
      }
    }
    lds_fence();

    // out_h = P @ V  (A from sq [q][key], B from svt [d][key] transposed)
#pragma unroll 1
    for (int ntd = 0; ntd < 2; ++ntd) {
      v16h bf = loadB(svt, TOKP, ntd * 16, 0, lane);
#pragma unroll
      for (int mt = 0; mt < 2; ++mt) {
        v16h a = loadA(sq, TOKP, mt * 16, 0, lane);
        v8f o = wmma16(a, bf, zero);
        storeD(ao, CH, mt * 16, h * HDIM + ntd * 16, lane, o);
      }
    }
    lds_fence();
  }

  // ---- projection: out = attn_out @ W_proj + b_proj (fp32 store) ----
  const size_t rowG = (size_t)b * TOK;
#pragma unroll 1
  for (int nt = 0; nt < CH / 16; ++nt) {
    v8f acc0 = zero, acc1 = zero;
#pragma unroll 1
    for (int ks = 0; ks < 8; ++ks) {
      v16h bf = loadB(wprojT, CH, nt * 16, ks * 32, lane);
      v16h a0 = loadA(ao, CH, 0, ks * 32, lane);
      v16h a1 = loadA(ao, CH, 16, ks * 32, lane);
      acc0 = wmma16(a0, bf, acc0);
      acc1 = wmma16(a1, bf, acc1);
    }
    const int c = nt * 16 + (lane & 15);
    const float bias = bpS[c];
    const int r0 = (lane >> 4) << 3;
#pragma unroll
    for (int rr = 0; rr < 8; ++rr) {
      int tok = r0 + rr;
      if (tok < TOK) out[(rowG + tok) * CH + c] = acc0[rr] + bias;
      tok += 16;
      if (tok < TOK) out[(rowG + tok) * CH + c] = acc1[rr] + bias;
    }
  }
}

// ---------------- launch ----------------

extern "C" void kernel_launch(void* const* d_in, const int* in_sizes, int n_in,
                              void* d_out, int out_size, void* d_ws,
                              size_t ws_size, hipStream_t stream) {
  const float* x     = (const float*)d_in[0];
  const float* adj   = (const float*)d_in[1];
  const float* Wqkv  = (const float*)d_in[2];
  const float* Wproj = (const float*)d_in[3];
  const float* bproj = (const float*)d_in[4];
  const float* Wgate = (const float*)d_in[5];
  const float* bgate = (const float*)d_in[6];
  float* out = (float*)d_out;

  h16* wqkvT  = (h16*)d_ws;                               // 768*256 f16
  h16* wprojT = (h16*)((char*)d_ws + 768 * 256 * 2);      // 256*256 f16
  h16* wgF    = (h16*)((char*)d_ws + 768 * 256 * 2 + 256 * 256 * 2);  // 256 f16

  const int prepN = 768 * 256 + 256 * 256 + 256;
  prep_weights<<<(prepN + 511) / 512, 512, 0, stream>>>(Wqkv, Wproj, Wgate,
                                                        wqkvT, wprojT, wgF);

  const int nBatch = in_sizes[0] / (TOK * CH);            // 16384
  fused_attn<<<nBatch / 4, 128, SM_BYTES, stream>>>(
      x, adj, bgate, bproj, wqkvT, wprojT, wgF, out);
}
